// VisionBDH_7713761264049
// MI455X (gfx1250) — matrix operand order repaired and checked by reading, not compiled, not run on hardware
//
#include <hip/hip_runtime.h>
#include <hip/hip_bf16.h>

// ---------------- model constants ----------------
#define B_   16
#define D_   768
#define NH_  12
#define N_   256
#define NL_  4
#define T_   576
#define NC_  1000
#define HN_  (NH_ * N_)      // 3072
#define KPATCH 768           // CIN*P*P = 3*16*16

typedef __bf16 bf16_t;
typedef __attribute__((ext_vector_type(16))) __bf16 bf16x16;
typedef __attribute__((ext_vector_type(8)))  float  floatx8;

// ---------------- GEMM tiling ----------------
#define TM 128
#define TN 128
#define TK 32
#define LK 40   // padded LDS row in bf16 elements (80B -> spreads banks, keeps 16B align)

#define EP_F32        0
#define EP_BF16       1
#define EP_RELU_BF16  2

// ---- CDNA5 async global->LDS path (guarded; falls back to reg staging) ----
#if defined(__HIP_DEVICE_COMPILE__) && __has_builtin(__builtin_amdgcn_global_load_async_to_lds_b128)
#define HAS_ASYNC_LDS 1
#else
#define HAS_ASYNC_LDS 0
#endif

#if HAS_ASYNC_LDS
// builtin signature (from compiler diagnostic): param1 = int4-vector pointer in AS(1)
typedef int v4i_t __attribute__((ext_vector_type(4)));
typedef __attribute__((address_space(1))) v4i_t gm_v4i;
typedef __attribute__((address_space(3))) v4i_t lds_v4i;

__device__ __forceinline__ void async_cp16(const bf16_t* g, bf16_t* l)
{
    // per-lane: LDS[l .. l+15] = MEM[g .. g+15], tracked by ASYNCcnt
    __builtin_amdgcn_global_load_async_to_lds_b128(
        (gm_v4i*)(unsigned long long)g,
        (lds_v4i*)(unsigned int)(unsigned long long)l,
        0, 0);
}

__device__ __forceinline__ void wait_async0()
{
#if __has_builtin(__builtin_amdgcn_s_wait_asynccnt)
    __builtin_amdgcn_s_wait_asynccnt(0);
#else
    asm volatile("s_wait_asynccnt 0x0" ::: "memory");
#endif
}
#endif

// =====================================================================
// Generic batched bf16 GEMM:  C[bz] = A[bz] (MxK, row-major, lda)
//                                   x Bt[bz]^T (Bt is Ncol x K row-major, ldb)
// 8 waves (wave32) per block; each wave owns a 32x64 patch of the 128x128 tile.
// WMMA: v_wmma_f32_16x16x32_bf16, f32 accumulation.
// =====================================================================
template<int EPI>
__global__ __launch_bounds__(256)
void gemm_bf16(const bf16_t* __restrict__ A,  long sA, int lda,
               const bf16_t* __restrict__ Bt, long sB, int ldb,
               void* __restrict__ Cv,         long sC, int ldc,
               int M, int Ncol, int K)
{
    __shared__ __align__(16) bf16_t As[2][TM][LK];
    __shared__ __align__(16) bf16_t Bs[2][TN][LK];

    const int tid  = threadIdx.x;
    const int lane = tid & 31;
    const int wave = tid >> 5;

    const long bz = blockIdx.z;
    const bf16_t* Ab = A  + bz * sA;
    const bf16_t* Bb = Bt + bz * sB;

    const int m0 = blockIdx.y * TM;
    const int n0 = blockIdx.x * TN;

    const int wm = (wave & 3) * 32;   // wave row offset in tile
    const int wn = (wave >> 2) * 64;  // wave col offset in tile

    floatx8 acc[2][4];
#pragma unroll
    for (int i = 0; i < 2; ++i)
#pragma unroll
        for (int j = 0; j < 4; ++j)
            acc[i][j] = (floatx8){0.f,0.f,0.f,0.f,0.f,0.f,0.f,0.f};

    const int KT = K / TK;

    // each thread moves 2x16B for A and 2x16B for B per k-tile
    int rowA[2], segA[2];
    bool okA[2], okB[2];
#pragma unroll
    for (int i = 0; i < 2; ++i) {
        int idx = tid + i * 256;
        rowA[i] = idx >> 2;
        segA[i] = (idx & 3) * 8;
        okA[i] = (m0 + rowA[i] < M);
        okB[i] = (n0 + rowA[i] < Ncol);
    }

    // ---- pre-zero OOB tile rows once (k-invariant); valid for both buffers ----
    {
        const uint4 z = make_uint4(0u, 0u, 0u, 0u);
#pragma unroll
        for (int i = 0; i < 2; ++i) {
            if (!okA[i]) {
                *(uint4*)&As[0][rowA[i]][segA[i]] = z;
                *(uint4*)&As[1][rowA[i]][segA[i]] = z;
            }
            if (!okB[i]) {
                *(uint4*)&Bs[0][rowA[i]][segA[i]] = z;
                *(uint4*)&Bs[1][rowA[i]][segA[i]] = z;
            }
        }
    }

#if HAS_ASYNC_LDS
    // ---- prologue: async DMA of tile 0 into LDS ----
#pragma unroll
    for (int i = 0; i < 2; ++i) {
        if (okA[i]) async_cp16(Ab + (long)(m0 + rowA[i]) * lda + segA[i], &As[0][rowA[i]][segA[i]]);
        if (okB[i]) async_cp16(Bb + (long)(n0 + rowA[i]) * ldb + segA[i], &Bs[0][rowA[i]][segA[i]]);
    }
    wait_async0();
    __syncthreads();
#else
    uint4 ra[2], rb[2];
#pragma unroll
    for (int i = 0; i < 2; ++i) {
        if (okA[i]) ra[i] = *(const uint4*)(Ab + (long)(m0 + rowA[i]) * lda + segA[i]);
        if (okB[i]) rb[i] = *(const uint4*)(Bb + (long)(n0 + rowA[i]) * ldb + segA[i]);
    }
#pragma unroll
    for (int i = 0; i < 2; ++i) {
        if (okA[i]) *(uint4*)&As[0][rowA[i]][segA[i]] = ra[i];
        if (okB[i]) *(uint4*)&Bs[0][rowA[i]][segA[i]] = rb[i];
    }
    __syncthreads();
#endif

    int buf = 0;
    for (int kt = 0; kt < KT; ++kt) {
        const bool more = (kt + 1 < KT);

#if HAS_ASYNC_LDS
        if (more) {
            const int k0 = (kt + 1) * TK;
#pragma unroll
            for (int i = 0; i < 2; ++i) {
                if (okA[i]) async_cp16(Ab + (long)(m0 + rowA[i]) * lda + k0 + segA[i],
                                       &As[buf ^ 1][rowA[i]][segA[i]]);
                if (okB[i]) async_cp16(Bb + (long)(n0 + rowA[i]) * ldb + k0 + segA[i],
                                       &Bs[buf ^ 1][rowA[i]][segA[i]]);
            }
        }
#else
        if (more) {
            const int k0 = (kt + 1) * TK;
#pragma unroll
            for (int i = 0; i < 2; ++i) {
                if (okA[i]) ra[i] = *(const uint4*)(Ab + (long)(m0 + rowA[i]) * lda + k0 + segA[i]);
                if (okB[i]) rb[i] = *(const uint4*)(Bb + (long)(n0 + rowA[i]) * ldb + k0 + segA[i]);
            }
            if (kt + 2 < KT) {  // L2 prefetch of the tile after next
                const int k2 = (kt + 2) * TK;
                if (okA[0]) __builtin_prefetch(Ab + (long)(m0 + rowA[0]) * lda + k2 + segA[0], 0, 1);
                if (okB[0]) __builtin_prefetch(Bb + (long)(n0 + rowA[0]) * ldb + k2 + segA[0], 0, 1);
            }
        }
#endif

        // ---- compute on current buffer ----
        {
            const int lr  = lane & 15;
            const int kga = (lane < 16) ? 0 : 8;    // A: K {kga..kga+7, kga+16..kga+23}
            const int kgb = (lane < 16) ? 0 : 8;    // B: K {kgb..kgb+7, kgb+16..kgb+23}
            union { bf16x16 v; uint4 u[2]; } af[2], bff[4];
#pragma unroll
            for (int i = 0; i < 2; ++i) {
                const bf16_t* p = &As[buf][wm + i * 16 + lr][0];
                af[i].u[0] = *(const uint4*)(p + kga);
                af[i].u[1] = *(const uint4*)(p + kga + 16);
            }
#pragma unroll
            for (int j = 0; j < 4; ++j) {
                const bf16_t* p = &Bs[buf][wn + j * 16 + lr][0];
                bff[j].u[0] = *(const uint4*)(p + kgb);
                bff[j].u[1] = *(const uint4*)(p + kgb + 16);
            }
#pragma unroll
            for (int i = 0; i < 2; ++i)
#pragma unroll
                for (int j = 0; j < 4; ++j)
                    acc[i][j] = __builtin_amdgcn_wmma_f32_16x16x32_bf16(
                        false, af[i].v, false, bff[j].v, (short)0, acc[i][j], false, false);
        }

        if (more) {
#if HAS_ASYNC_LDS
            wait_async0();       // our DMA into buf^1 done; barrier orders all waves
            __syncthreads();
#else
            // write-back of next tile: buffer buf^1 was last READ at iteration kt-1,
            // which is already fenced by that iteration's barrier -> single barrier.
#pragma unroll
            for (int i = 0; i < 2; ++i) {
                if (okA[i]) *(uint4*)&As[buf ^ 1][rowA[i]][segA[i]] = ra[i];
                if (okB[i]) *(uint4*)&Bs[buf ^ 1][rowA[i]][segA[i]] = rb[i];
            }
            __syncthreads();
#endif
            buf ^= 1;
        }
    }

    // ---- epilogue: C/D layout: n = lane&15, m = r + 8*(lane>=16) ----
    const int cn  = lane & 15;
    const int cr8 = (lane >> 4) * 8;
#pragma unroll
    for (int i = 0; i < 2; ++i) {
#pragma unroll
        for (int j = 0; j < 4; ++j) {
            const int nn = n0 + wn + j * 16 + cn;
            if (nn >= Ncol) continue;
            const int mb = m0 + wm + i * 16 + cr8;
#pragma unroll
            for (int r = 0; r < 8; ++r) {
                const int mm = mb + r;
                if (mm >= M) continue;
                const float v = acc[i][j][r];
                if (EPI == EP_F32) {
                    ((float*)Cv)[bz * sC + (long)mm * ldc + nn] = v;
                } else if (EPI == EP_BF16) {
                    ((bf16_t*)Cv)[bz * sC + (long)mm * ldc + nn] = (bf16_t)v;
                } else {
                    ((bf16_t*)Cv)[bz * sC + (long)mm * ldc + nn] = (bf16_t)fmaxf(v, 0.f);
                }
            }
        }
    }
}

// =====================================================================
// Elementwise / reduction kernels
// =====================================================================
__global__ void convert_bf16_kernel(const float* __restrict__ in, bf16_t* __restrict__ out, long n)
{
    long i = (long)blockIdx.x * 256 + threadIdx.x;
    if (i < n) out[i] = (bf16_t)in[i];
}

// dst[b][c][r] = src[b][r][c]  (f32 -> bf16)
__global__ void transpose_pack_kernel(const float* __restrict__ src, bf16_t* __restrict__ dst,
                                      int R, int C, long n)
{
    long i = (long)blockIdx.x * 256 + threadIdx.x;
    if (i >= n) return;
    int c = (int)(i % C);
    long t = i / C;
    int r = (int)(t % R);
    long b = t / R;
    dst[(b * C + c) * (long)R + r] = (bf16_t)src[i];
}

// h f32 [B][T][D] -> hb bf16 (same layout) and hTb bf16 [B][D][T]
__global__ void pack_h_kernel(const float* __restrict__ h, bf16_t* __restrict__ hb,
                              bf16_t* __restrict__ hTb)
{
    long i = (long)blockIdx.x * 256 + threadIdx.x;
    if (i >= (long)B_ * T_ * D_) return;
    int d = (int)(i % D_);
    long t2 = i / D_;
    int t = (int)(t2 % T_);
    int b = (int)(t2 / T_);
    bf16_t v = (bf16_t)h[i];
    hb[i] = v;
    hTb[((long)b * D_ + d) * T_ + t] = v;
}

// im2col for 16x16 stride-16 patches: patches[b*T+t][k] with k = c*256 + py*16 + px
__global__ void im2col_kernel(const float* __restrict__ x, bf16_t* __restrict__ patches)
{
    long i = (long)blockIdx.x * 256 + threadIdx.x;
    if (i >= (long)B_ * T_ * KPATCH) return;
    int k = (int)(i % KPATCH);
    long bt = i / KPATCH;
    int t = (int)(bt % T_);
    int b = (int)(bt / T_);
    int c   = k >> 8;
    int rem = k & 255;
    int py = rem >> 4, px = rem & 15;
    int ty = t / 24, tx = t % 24;               // IMG/P = 24
    long src = (((long)(b * 3 + c) * 384) + ty * 16 + py) * 384 + tx * 16 + px;
    patches[i] = (bf16_t)x[src];
}

__device__ inline void block_reduce2(float& s, float& q, float* sh1, float* sh2)
{
    const int tid = threadIdx.x;
    sh1[tid] = s; sh2[tid] = q;
    __syncthreads();
    for (int off = 128; off > 0; off >>= 1) {
        if (tid < off) { sh1[tid] += sh1[tid + off]; sh2[tid] += sh2[tid + off]; }
        __syncthreads();
    }
    s = sh1[0]; q = sh2[0];
    __syncthreads();
}

// out = LN(in), row width 768, one block per row
template<bool OBF>
__global__ __launch_bounds__(256)
void ln_rows(const float* __restrict__ in, void* __restrict__ out)
{
    __shared__ float sh1[256], sh2[256];
    const int tid = threadIdx.x;
    const long row = blockIdx.x;
    const float* p = in + row * D_;
    float v[3]; float s = 0.f, q = 0.f;
#pragma unroll
    for (int i = 0; i < 3; ++i) { v[i] = p[tid + i * 256]; s += v[i]; q += v[i] * v[i]; }
    block_reduce2(s, q, sh1, sh2);
    const float mean = s * (1.f / 768.f);
    const float var  = q * (1.f / 768.f) - mean * mean;
    const float r    = rsqrtf(var + 1e-5f);
#pragma unroll
    for (int i = 0; i < 3; ++i) {
        float o = (v[i] - mean) * r;
        if (OBF) ((bf16_t*)out)[row * D_ + tid + i * 256] = (bf16_t)o;
        else     ((float*)out)[row * D_ + tid + i * 256]  = o;
    }
}

// h = LN(raw + conv_b + pos_embed[t])
__global__ __launch_bounds__(256)
void ln_bias_pos(const float* __restrict__ raw, const float* __restrict__ cb,
                 const float* __restrict__ pos, float* __restrict__ h)
{
    __shared__ float sh1[256], sh2[256];
    const int tid = threadIdx.x;
    const long row = blockIdx.x;        // b*T + t
    const int t = (int)(row % T_);
    const float* p = raw + row * D_;
    float v[3]; float s = 0.f, q = 0.f;
#pragma unroll
    for (int i = 0; i < 3; ++i) {
        int d = tid + i * 256;
        v[i] = p[d] + cb[d] + pos[(long)t * D_ + d];
        s += v[i]; q += v[i] * v[i];
    }
    block_reduce2(s, q, sh1, sh2);
    const float mean = s * (1.f / 768.f);
    const float var  = q * (1.f / 768.f) - mean * mean;
    const float r    = rsqrtf(var + 1e-5f);
#pragma unroll
    for (int i = 0; i < 3; ++i) h[row * D_ + tid + i * 256] = (v[i] - mean) * r;
}

// h = LN(h + LN(raw))
__global__ __launch_bounds__(256)
void resid_ln(const float* __restrict__ raw, float* __restrict__ h)
{
    __shared__ float sh1[256], sh2[256];
    const int tid = threadIdx.x;
    const long row = blockIdx.x;
    const float* p = raw + row * D_;
    float* ph = h + row * D_;
    float v[3], u[3]; float s = 0.f, q = 0.f;
#pragma unroll
    for (int i = 0; i < 3; ++i) { v[i] = p[tid + i * 256]; s += v[i]; q += v[i] * v[i]; }
    block_reduce2(s, q, sh1, sh2);
    const float m1 = s * (1.f / 768.f);
    const float r1 = rsqrtf(q * (1.f / 768.f) - m1 * m1 + 1e-5f);
    float s2 = 0.f, q2 = 0.f;
#pragma unroll
    for (int i = 0; i < 3; ++i) {
        u[i] = ph[tid + i * 256] + (v[i] - m1) * r1;
        s2 += u[i]; q2 += u[i] * u[i];
    }
    block_reduce2(s2, q2, sh1, sh2);
    const float m2 = s2 * (1.f / 768.f);
    const float r2 = rsqrtf(q2 * (1.f / 768.f) - m2 * m2 + 1e-5f);
#pragma unroll
    for (int i = 0; i < 3; ++i) ph[tid + i * 256] = (u[i] - m2) * r2;
}

// qr = rope(x_sp); layouts [b][t][h*256+n]; freq(n) = 2^(-(n>>1)/8) (THETA = 2^16)
__global__ void rope_kernel(const bf16_t* __restrict__ xsp, bf16_t* __restrict__ qr)
{
    long i = (long)blockIdx.x * 256 + threadIdx.x;          // pair index
    if (i >= (long)B_ * T_ * HN_ / 2) return;
    long e = i * 2;
    int hn = (int)(e % HN_);
    long bt = e / HN_;
    int t = (int)(bt % T_);
    int n = hn & 255;
    float f  = exp2f(-(float)(n >> 1) * 0.125f);
    float ph = (float)t * f;
    ph -= floorf(ph);
    float ang = ph * 6.28318530717958647692f;
    float sn, cs;
    __sincosf(ang, &sn, &cs);
    float x0 = (float)xsp[e], x1 = (float)xsp[e + 1];
    qr[e]     = (bf16_t)(x0 * cs - x1 * sn);
    qr[e + 1] = (bf16_t)(x1 * cs + x0 * sn);
}

// xsp[b] (as xy) *= ysp_b ; xsp layout [t][h*256+n], ysp layout [h][t][n]
__global__ void xy_mul_kernel(bf16_t* __restrict__ xsp_b, const bf16_t* __restrict__ ysp_b)
{
    long i = (long)blockIdx.x * 256 + threadIdx.x;
    if (i >= (long)T_ * HN_) return;
    int hn = (int)(i % HN_);
    int t  = (int)(i / HN_);
    int hh = hn >> 8, n = hn & 255;
    float a = (float)xsp_b[i];
    float v = (float)ysp_b[((long)hh * T_ + t) * N_ + n];
    xsp_b[i] = (bf16_t)(a * v);
}

// pooled_raw[b][d] = mean_t h[b][t][d]
__global__ void pool_mean_kernel(const float* __restrict__ h, float* __restrict__ pr)
{
    int i = blockIdx.x * 256 + threadIdx.x;
    if (i >= B_ * D_) return;
    int d = i % D_, b = i / D_;
    float s = 0.f;
    for (int t = 0; t < T_; ++t) s += h[((long)b * T_ + t) * D_ + d];
    pr[i] = s * (1.f / (float)T_);
}

__global__ void head_bias_kernel(const float* __restrict__ hr, const float* __restrict__ hb,
                                 float* __restrict__ out)
{
    int i = blockIdx.x * 256 + threadIdx.x;
    if (i >= B_ * NC_) return;
    out[i] = hr[i] + hb[i % NC_];
}

// =====================================================================
// Host launcher
// =====================================================================
extern "C" void kernel_launch(void* const* d_in, const int* in_sizes, int n_in,
                              void* d_out, int out_size, void* d_ws, size_t ws_size,
                              hipStream_t stream)
{
    (void)in_sizes; (void)n_in; (void)out_size; (void)ws_size;

    const float* x      = (const float*)d_in[0];
    const float* conv_w = (const float*)d_in[1];
    const float* conv_b = (const float*)d_in[2];
    const float* pos    = (const float*)d_in[3];
    const float* enc    = (const float*)d_in[4];
    const float* encv   = (const float*)d_in[5];
    const float* dec    = (const float*)d_in[6];
    const float* hw     = (const float*)d_in[7];
    const float* hbias  = (const float*)d_in[8];
    float* out = (float*)d_out;

    char* wsb = (char*)d_ws;
    size_t off = 0;
    auto carve = [&](size_t bytes) -> char* {
        char* p = wsb + off;
        off += (bytes + 255) & ~(size_t)255;
        return p;
    };

    float*  h      = (float*) carve(sizeof(float) * (size_t)B_ * T_ * D_);
    float*  raw    = (float*) carve(sizeof(float) * (size_t)B_ * T_ * D_);   // conv raw / ymlp raw
    bf16_t* hb     = (bf16_t*)carve(2ull * B_ * T_ * D_);
    bf16_t* hTb    = (bf16_t*)carve(2ull * B_ * T_ * D_);
    bf16_t* xsp    = (bf16_t*)carve(2ull * B_ * T_ * HN_);                   // x_sp, later xy (in-place)
    bf16_t* qr     = (bf16_t*)carve(2ull * B_ * T_ * HN_);
    bf16_t* sc     = (bf16_t*)carve(2ull * NH_ * T_ * T_);                   // per-b scores
    float*  ykvraw = (float*) carve(4ull * NH_ * T_ * D_);                   // per-b
    bf16_t* ykv    = (bf16_t*)carve(2ull * NH_ * T_ * D_);                   // per-b
    bf16_t* ysp    = (bf16_t*)carve(2ull * NH_ * T_ * N_);                   // per-b
    bf16_t* pat    = (bf16_t*)carve(2ull * B_ * T_ * KPATCH);
    bf16_t* encT   = (bf16_t*)carve(2ull * NH_ * D_ * N_);                   // [h][n][d]
    bf16_t* encVT  = (bf16_t*)carve(2ull * NH_ * D_ * N_);
    bf16_t* decT   = (bf16_t*)carve(2ull * D_ * HN_);                        // [d][hn]
    bf16_t* cwb    = (bf16_t*)carve(2ull * D_ * KPATCH);
    bf16_t* hwb    = (bf16_t*)carve(2ull * NC_ * D_);
    float*  poolr  = (float*) carve(4ull * B_ * D_);
    bf16_t* poolb  = (bf16_t*)carve(2ull * B_ * D_);
    float*  hraw   = (float*) carve(4ull * B_ * NC_);

    dim3 blk(256);
    auto g1 = [](long n) { return dim3((unsigned)((n + 255) / 256)); };

    // ---- weight packing (once per call; cheap) ----
    convert_bf16_kernel<<<g1((long)D_ * KPATCH), blk, 0, stream>>>(conv_w, cwb, (long)D_ * KPATCH);
    convert_bf16_kernel<<<g1((long)NC_ * D_),    blk, 0, stream>>>(hw,     hwb, (long)NC_ * D_);
    transpose_pack_kernel<<<g1((long)NH_ * D_ * N_), blk, 0, stream>>>(enc,  encT,  D_,  N_, (long)NH_ * D_ * N_);
    transpose_pack_kernel<<<g1((long)NH_ * D_ * N_), blk, 0, stream>>>(encv, encVT, D_,  N_, (long)NH_ * D_ * N_);
    transpose_pack_kernel<<<g1((long)HN_ * D_),      blk, 0, stream>>>(dec,  decT,  HN_, D_, (long)HN_ * D_);

    // ---- patch embed: im2col + GEMM + bias/pos + LN ----
    im2col_kernel<<<g1((long)B_ * T_ * KPATCH), blk, 0, stream>>>(x, pat);
    gemm_bf16<EP_F32><<<dim3(6, 72, 1), blk, 0, stream>>>(
        pat, 0, KPATCH, cwb, 0, KPATCH, raw, 0, D_, B_ * T_, D_, KPATCH);
    ln_bias_pos<<<dim3(B_ * T_), blk, 0, stream>>>(raw, conv_b, pos, h);

    // ---- BDH layers ----
    for (int L = 0; L < NL_; ++L) {
        pack_h_kernel<<<g1((long)B_ * T_ * D_), blk, 0, stream>>>(h, hb, hTb);

        // x_sp = relu(h @ enc)  : [B](576 x 3072 x 768)
        gemm_bf16<EP_RELU_BF16><<<dim3(24, 5, B_), blk, 0, stream>>>(
            hb, (long)T_ * D_, D_, encT, 0, D_, xsp, (long)T_ * HN_, HN_, T_, HN_, D_);

        rope_kernel<<<g1((long)B_ * T_ * HN_ / 2), blk, 0, stream>>>(xsp, qr);

        for (int b = 0; b < B_; ++b) {
            const bf16_t* qrb = qr + (long)b * T_ * HN_;
            // scores = qr @ qr^T : [NH](576 x 576 x 256)
            gemm_bf16<EP_BF16><<<dim3(5, 5, NH_), blk, 0, stream>>>(
                qrb, 256, HN_, qrb, 256, HN_, sc, (long)T_ * T_, T_, T_, T_, N_);
            // ykv_raw = scores @ h : [NH](576 x 768 x 576)
            gemm_bf16<EP_F32><<<dim3(6, 5, NH_), blk, 0, stream>>>(
                sc, (long)T_ * T_, T_, hTb + (long)b * D_ * T_, 0, T_,
                ykvraw, (long)T_ * D_, D_, T_, D_, T_);
            // ykv = LN(ykv_raw)
            ln_rows<true><<<dim3(NH_ * T_), blk, 0, stream>>>(ykvraw, ykv);
            // y_sp = relu(ykv @ encV) : [NH](576 x 256 x 768)
            gemm_bf16<EP_RELU_BF16><<<dim3(2, 5, NH_), blk, 0, stream>>>(
                ykv, (long)T_ * D_, D_, encVT, (long)N_ * D_, D_,
                ysp, (long)T_ * N_, N_, T_, N_, D_);
            // xy = x_sp * y_sp (in place into xsp)
            xy_mul_kernel<<<g1((long)T_ * HN_), blk, 0, stream>>>(xsp + (long)b * T_ * HN_, ysp);
        }

        // ymlp_raw = xy @ dec : [B](576 x 768 x 3072)
        gemm_bf16<EP_F32><<<dim3(6, 5, B_), blk, 0, stream>>>(
            xsp, (long)T_ * HN_, HN_, decT, 0, HN_, raw, (long)T_ * D_, D_, T_, D_, HN_);

        // h = LN(h + LN(ymlp))
        resid_ln<<<dim3(B_ * T_), blk, 0, stream>>>(raw, h);
    }

    // ---- head ----
    pool_mean_kernel<<<g1(B_ * D_), blk, 0, stream>>>(h, poolr);
    ln_rows<true><<<dim3(B_), blk, 0, stream>>>(poolr, poolb);
    gemm_bf16<EP_F32><<<dim3(8, 1, 1), blk, 0, stream>>>(
        poolb, 0, D_, hwb, 0, D_, hraw, 0, NC_, B_, NC_, D_);
    head_bias_kernel<<<g1(B_ * NC_), blk, 0, stream>>>(hraw, hbias, out);
}
